// SemiHardTripletLoss_48198122996409
// MI455X (gfx1250) — matrix operand order repaired and checked by reading, not compiled, not run on hardware
//
#include <hip/hip_runtime.h>

typedef __attribute__((ext_vector_type(16))) _Float16 v16h;
typedef __attribute__((ext_vector_type(8)))  _Float16 v8h;
typedef __attribute__((ext_vector_type(4)))  _Float16 v4h;
typedef __attribute__((ext_vector_type(8)))  float    v8f;

#define D_DIM 128
#define KF 4             // D_DIM / 32 K-fragments per WMMA tile
#define SPLIT 4          // column splits (parallelism across blocks)
#define MARGIN_F 0.2f
#define INF_F 1e30f

// LDS panel: 32 columns x 128 halves, each column padded +8 halves (16B)
#define COL_STRIDE 136                    // halves per column row in LDS
#define PANEL_COLS 32                     // columns staged per buffer (2 tiles)
#define MAT_HALVES (PANEL_COLS * COL_STRIDE)  // one matrix (hi or lo)
#define BUF_HALVES (2 * MAT_HALVES)           // hi + lo

#define USE_ASYNC_LDS 1

__device__ __forceinline__ void async_wait0() {
#if USE_ASYNC_LDS
#if __has_builtin(__builtin_amdgcn_s_wait_asynccnt)
  __builtin_amdgcn_s_wait_asynccnt(0);
#else
  asm volatile("s_wait_asynccnt 0x0" ::: "memory");
#endif
#endif
}

// 16-byte global -> LDS transfer (async on CDNA5, sync fallback)
__device__ __forceinline__ void copy16_to_lds(void* l, const void* g) {
#if USE_ASYNC_LDS
  const unsigned lds_off = (unsigned)(size_t)l;
  const unsigned long long ga = (unsigned long long)(size_t)g;
  asm volatile("global_load_async_to_lds_b128 %0, %1, off"
               :: "v"(lds_off), "v"(ga) : "memory");
#else
  *(uint4*)l = *(const uint4*)g;
#endif
}

// ---- order-preserving float <-> uint for atomic min/max ----
__device__ __forceinline__ unsigned fkey(float f) {
  unsigned u = __float_as_uint(f);
  return (u & 0x80000000u) ? ~u : (u | 0x80000000u);
}
__device__ __forceinline__ float fdec(unsigned k) {
  unsigned u = (k & 0x80000000u) ? (k ^ 0x80000000u) : ~k;
  return __uint_as_float(u);
}

// ---------------------------------------------------------------------------
__global__ void init_kernel(unsigned* dap, unsigned* nmin, unsigned* smin,
                            unsigned* flags, float* sum, unsigned* cnt, int B) {
  int i = blockIdx.x * blockDim.x + threadIdx.x;
  if (i < B) {
    dap[i]   = 0u;           // lowest key
    nmin[i]  = 0xFFFFFFFFu;  // highest key
    smin[i]  = 0xFFFFFFFFu;
    flags[i] = 0u;
  }
  if (i == 0) { *sum = 0.0f; *cnt = 0u; }
}

// ---------------------------------------------------------------------------
// One wave per row: f32 normalize, then split into hi (f16) + lo (residual f16)
__global__ void __launch_bounds__(256)
normalize_kernel(const float* __restrict__ x,
                 _Float16* __restrict__ xhi, _Float16* __restrict__ xlo, int B) {
  int row  = (blockIdx.x * blockDim.x + threadIdx.x) >> 5;
  int lane = threadIdx.x & 31;
  if (row >= B) return;
  const float4 v = ((const float4*)(x + (size_t)row * D_DIM))[lane];
  float ss = v.x * v.x + v.y * v.y + v.z * v.z + v.w * v.w;
  #pragma unroll
  for (int m = 1; m < 32; m <<= 1) ss += __shfl_xor(ss, m, 32);
  const float r = 1.0f / sqrtf(ss);
  float y[4] = { v.x * r, v.y * r, v.z * r, v.w * r };
  v4h hi, lo;
  #pragma unroll
  for (int c = 0; c < 4; c++) {
    _Float16 h = (_Float16)y[c];
    hi[c] = h;
    lo[c] = (_Float16)(y[c] - (float)h);
  }
  *(v4h*)(xhi + (size_t)row * D_DIM + lane * 4) = hi;
  *(v4h*)(xlo + (size_t)row * D_DIM + lane * 4) = lo;
}

// ---------------------------------------------------------------------------
// Stage one 32-column panel (hi+lo) + its 32 labels into LDS.
// 1024 x 16B matrix transfers (4/thread) + 8 x 16B label transfers.
__device__ __forceinline__ void stage_panel(const _Float16* __restrict__ xhi,
                                            const _Float16* __restrict__ xlo,
                                            const int* __restrict__ labels,
                                            _Float16* sbuf, int* slab,
                                            int jbase, int tid) {
  #pragma unroll
  for (int s = 0; s < 4; s++) {
    const int u  = tid + s * 256;
    const int m  = u >> 9;         // 0 = hi, 1 = lo
    const int c  = (u >> 4) & 31;  // column within panel
    const int ch = u & 15;         // 16-byte chunk within 256B row
    const _Float16* g = (m ? xlo : xhi) + (size_t)(jbase + c) * D_DIM + ch * 8;
    _Float16* l = sbuf + m * MAT_HALVES + c * COL_STRIDE + ch * 8;
    copy16_to_lds(l, g);
  }
  if (tid < 8) copy16_to_lds(&slab[tid * 4], &labels[jbase + tid * 4]);
}

// ---------------------------------------------------------------------------
// Load the wave's A fragments (16 anchors x 32 K per fragment), per ISA layout:
// lane L: row M = L%16; VGPRs 0..3 hold K = (L/16)*8 + 0..7, VGPRs 4..7 hold +16.
__device__ __forceinline__ void load_a_frags(const _Float16* __restrict__ xhi,
                                             const _Float16* __restrict__ xlo,
                                             int ibase, int h, int n,
                                             v16h* ahi, v16h* alo) {
  const _Float16* rh = xhi + (size_t)(ibase + n) * D_DIM;
  const _Float16* rl = xlo + (size_t)(ibase + n) * D_DIM;
  #pragma unroll
  for (int f = 0; f < KF; f++) {
    const int k0 = f * 32 + h * 8;
    const int k1 = f * 32 + 16 + h * 8;
    union { v16h v; v8h p[2]; } uh, ul;
    uh.p[0] = *(const v8h*)(rh + k0);
    uh.p[1] = *(const v8h*)(rh + k1);
    ul.p[0] = *(const v8h*)(rl + k0);
    ul.p[1] = *(const v8h*)(rl + k1);
    ahi[f] = uh.v;
    alo[f] = ul.v;
  }
}

// 16x16 Gram tile t (0/1) of the staged 32-col panel. All 8 B fragments are
// preloaded (single DScnt wait), then two independent WMMA chains.
// B fragment: lane L: col N = L%16; K = (L/16)*8 + 0..7, then +16.
__device__ __forceinline__ v8f gram_tile_lds(const _Float16* sbuf,
                                             const v16h* ahi, const v16h* alo,
                                             int h, int n, int t) {
  const _Float16* bhp = sbuf + (t * 16 + n) * COL_STRIDE;
  const _Float16* blp = bhp + MAT_HALVES;
  v16h bh[KF], bl[KF];
  #pragma unroll
  for (int f = 0; f < KF; f++) {
    const int off = f * 32 + h * 8;
    union { v16h v; v8h p[2]; } ub, ul;
    ub.p[0] = *(const v8h*)(bhp + off);
    ub.p[1] = *(const v8h*)(bhp + off + 16);
    ul.p[0] = *(const v8h*)(blp + off);
    ul.p[1] = *(const v8h*)(blp + off + 16);
    bh[f] = ub.v;
    bl[f] = ul.v;
  }
  v8f acc0 = {}, acc1 = {};
  #pragma unroll
  for (int f = 0; f < KF; f++) {
    acc0 = __builtin_amdgcn_wmma_f32_16x16x32_f16(false, ahi[f], false, bh[f],
                                                  (short)0, acc0, false, false);
    acc1 = __builtin_amdgcn_wmma_f32_16x16x32_f16(false, alo[f], false, bh[f],
                                                  (short)0, acc1, false, false);
    acc1 = __builtin_amdgcn_wmma_f32_16x16x32_f16(false, ahi[f], false, bl[f],
                                                  (short)0, acc1, false, false);
  }
  return acc0 + acc1;
}

// ---------------------------------------------------------------------------
// Pass 1: hardest positive (max dist, same label, i!=j), min negative dist,
// has_pos / has_neg flags. Block = 8 waves x 16 anchors, shared column split.
__global__ void __launch_bounds__(256)
pass1_kernel(const _Float16* __restrict__ xhi, const _Float16* __restrict__ xlo,
             const int* __restrict__ labels,
             unsigned* __restrict__ dap_key, unsigned* __restrict__ nmin_key,
             unsigned* __restrict__ flags, int B) {
  __shared__ __align__(32) _Float16 sbuf[2][BUF_HALVES];
  __shared__ __align__(16) int slab[2][PANEL_COLS];
  const int tid   = threadIdx.x;
  const int lane  = tid & 31;
  const int wave  = tid >> 5;
  const int split = blockIdx.x % SPLIT;
  const int ibase = (blockIdx.x / SPLIT) * 128 + wave * 16;
  const int h = lane >> 4, n = lane & 15;

  int labr[8];
  #pragma unroll
  for (int r = 0; r < 8; r++) labr[r] = labels[ibase + h * 8 + r];

  v16h ahi[KF], alo[KF];
  load_a_frags(xhi, xlo, ibase, h, n, ahi, alo);

  float pmax[8], nmin[8];
  int panym = 0, nanym = 0;
  #pragma unroll
  for (int r = 0; r < 8; r++) { pmax[r] = -INF_F; nmin[r] = INF_F; }

  const int ctiles = (B / PANEL_COLS) / SPLIT;
  const int jt0 = split * ctiles;

  stage_panel(xhi, xlo, labels, sbuf[0], slab[0], jt0 * PANEL_COLS, tid);
  async_wait0();
  __syncthreads();

  for (int jt = 0; jt < ctiles; jt++) {
    const int cur = jt & 1;
    if (jt + 1 < ctiles)
      stage_panel(xhi, xlo, labels, sbuf[1 ^ cur], slab[1 ^ cur],
                  (jt0 + jt + 1) * PANEL_COLS, tid);

    const int jbase = (jt0 + jt) * PANEL_COLS;
    #pragma unroll
    for (int t = 0; t < 2; t++) {
      const int j = jbase + t * 16 + n;
      const int labc = slab[cur][t * 16 + n];
      const v8f acc = gram_tile_lds(sbuf[cur], ahi, alo, h, n, t);
      #pragma unroll
      for (int r = 0; r < 8; r++) {
        const int i = ibase + h * 8 + r;
        const float d = 1.0f - acc[r];
        const bool same = (labr[r] == labc);
        if (same && (i != j)) { pmax[r] = fmaxf(pmax[r], d); panym |= 1 << r; }
        if (!same)            { nmin[r] = fminf(nmin[r], d); nanym |= 1 << r; }
      }
    }
    if (jt + 1 < ctiles) async_wait0();
    __syncthreads();
  }

  // reduce across the 16 lanes (columns) of each half-wave
  #pragma unroll
  for (int r = 0; r < 8; r++) {
    #pragma unroll
    for (int m = 1; m < 16; m <<= 1) {
      pmax[r] = fmaxf(pmax[r], __shfl_xor(pmax[r], m, 32));
      nmin[r] = fminf(nmin[r], __shfl_xor(nmin[r], m, 32));
    }
  }
  #pragma unroll
  for (int m = 1; m < 16; m <<= 1) {
    panym |= __shfl_xor(panym, m, 32);
    nanym |= __shfl_xor(nanym, m, 32);
  }

  if (n == 0) {
    #pragma unroll
    for (int r = 0; r < 8; r++) {
      const int i = ibase + h * 8 + r;
      if ((panym >> r) & 1) atomicMax(&dap_key[i], fkey(pmax[r]));
      if ((nanym >> r) & 1) atomicMin(&nmin_key[i], fkey(nmin[r]));
      const unsigned fl = (unsigned)(((panym >> r) & 1) | (((nanym >> r) & 1) << 1));
      if (fl) atomicOr(&flags[i], fl);
    }
  }
}

// ---------------------------------------------------------------------------
// Pass 2: semi-hard negative = min over negatives with dist > d_ap.
__global__ void __launch_bounds__(256)
pass2_kernel(const _Float16* __restrict__ xhi, const _Float16* __restrict__ xlo,
             const int* __restrict__ labels,
             const unsigned* __restrict__ dap_key,
             unsigned* __restrict__ smin_key, unsigned* __restrict__ flags, int B) {
  __shared__ __align__(32) _Float16 sbuf[2][BUF_HALVES];
  __shared__ __align__(16) int slab[2][PANEL_COLS];
  const int tid   = threadIdx.x;
  const int lane  = tid & 31;
  const int wave  = tid >> 5;
  const int split = blockIdx.x % SPLIT;
  const int ibase = (blockIdx.x / SPLIT) * 128 + wave * 16;
  const int h = lane >> 4, n = lane & 15;

  int labr[8];
  float dapr[8];
  #pragma unroll
  for (int r = 0; r < 8; r++) {
    labr[r] = labels[ibase + h * 8 + r];
    dapr[r] = fdec(dap_key[ibase + h * 8 + r]);
  }

  v16h ahi[KF], alo[KF];
  load_a_frags(xhi, xlo, ibase, h, n, ahi, alo);

  float smin[8];
  int sanym = 0;
  #pragma unroll
  for (int r = 0; r < 8; r++) smin[r] = INF_F;

  const int ctiles = (B / PANEL_COLS) / SPLIT;
  const int jt0 = split * ctiles;

  stage_panel(xhi, xlo, labels, sbuf[0], slab[0], jt0 * PANEL_COLS, tid);
  async_wait0();
  __syncthreads();

  for (int jt = 0; jt < ctiles; jt++) {
    const int cur = jt & 1;
    if (jt + 1 < ctiles)
      stage_panel(xhi, xlo, labels, sbuf[1 ^ cur], slab[1 ^ cur],
                  (jt0 + jt + 1) * PANEL_COLS, tid);

    #pragma unroll
    for (int t = 0; t < 2; t++) {
      const int labc = slab[cur][t * 16 + n];
      const v8f acc = gram_tile_lds(sbuf[cur], ahi, alo, h, n, t);
      #pragma unroll
      for (int r = 0; r < 8; r++) {
        const float d = 1.0f - acc[r];
        if ((labr[r] != labc) && (d > dapr[r])) {
          smin[r] = fminf(smin[r], d);
          sanym |= 1 << r;
        }
      }
    }
    if (jt + 1 < ctiles) async_wait0();
    __syncthreads();
  }

  #pragma unroll
  for (int r = 0; r < 8; r++) {
    #pragma unroll
    for (int m = 1; m < 16; m <<= 1)
      smin[r] = fminf(smin[r], __shfl_xor(smin[r], m, 32));
  }
  #pragma unroll
  for (int m = 1; m < 16; m <<= 1) sanym |= __shfl_xor(sanym, m, 32);

  if (n == 0) {
    #pragma unroll
    for (int r = 0; r < 8; r++) {
      const int i = ibase + h * 8 + r;
      if ((sanym >> r) & 1) {
        atomicMin(&smin_key[i], fkey(smin[r]));
        atomicOr(&flags[i], 4u);
      }
    }
  }
}

// ---------------------------------------------------------------------------
__global__ void __launch_bounds__(256)
finalize_kernel(const unsigned* __restrict__ dap, const unsigned* __restrict__ nmin,
                const unsigned* __restrict__ smin, const unsigned* __restrict__ flags,
                float* __restrict__ sum, unsigned* __restrict__ cnt, int B) {
  const int i = blockIdx.x * blockDim.x + threadIdx.x;
  const int lane = threadIdx.x & 31;
  const int wave = threadIdx.x >> 5;
  float v = 0.0f;
  unsigned c = 0u;
  if (i < B) {
    const unsigned fl = flags[i];
    if ((fl & 3u) == 3u) {  // has positive & has negative
      const float d_ap = fdec(dap[i]);
      const float d_an = (fl & 4u) ? fdec(smin[i]) : fdec(nmin[i]);
      v = fmaxf(d_ap - d_an + MARGIN_F, 0.0f);
      c = 1u;
    }
  }
  #pragma unroll
  for (int m = 1; m < 32; m <<= 1) {
    v += __shfl_xor(v, m, 32);
    c += __shfl_xor(c, m, 32);
  }
  __shared__ float sv[8];
  __shared__ unsigned sc[8];
  if (lane == 0) { sv[wave] = v; sc[wave] = c; }
  __syncthreads();
  if (threadIdx.x == 0) {
    float tv = 0.0f; unsigned tc = 0u;
    #pragma unroll
    for (int w = 0; w < 8; w++) { tv += sv[w]; tc += sc[w]; }
    if (tv != 0.0f || tc != 0u) { atomicAdd(sum, tv); atomicAdd(cnt, tc); }
  }
}

__global__ void writeout_kernel(const float* __restrict__ sum,
                                const unsigned* __restrict__ cnt,
                                float* __restrict__ out) {
  if (threadIdx.x == 0 && blockIdx.x == 0) {
    const unsigned c = *cnt;
    out[0] = *sum / (float)(c > 0u ? c : 1u);
  }
}

// ---------------------------------------------------------------------------
extern "C" void kernel_launch(void* const* d_in, const int* in_sizes, int n_in,
                              void* d_out, int out_size, void* d_ws, size_t ws_size,
                              hipStream_t stream) {
  const float* emb   = (const float*)d_in[0];
  const int* labels  = (const int*)d_in[1];
  const int B        = in_sizes[1];           // 8192 anchors
  float* out         = (float*)d_out;

  char* ws = (char*)d_ws;
  size_t off = 0;
  _Float16* xhi = (_Float16*)(ws + off); off += (size_t)B * D_DIM * sizeof(_Float16);
  _Float16* xlo = (_Float16*)(ws + off); off += (size_t)B * D_DIM * sizeof(_Float16);
  unsigned* dap  = (unsigned*)(ws + off); off += (size_t)B * sizeof(unsigned);
  unsigned* nmin = (unsigned*)(ws + off); off += (size_t)B * sizeof(unsigned);
  unsigned* smin = (unsigned*)(ws + off); off += (size_t)B * sizeof(unsigned);
  unsigned* flags= (unsigned*)(ws + off); off += (size_t)B * sizeof(unsigned);
  float*    sum  = (float*)   (ws + off); off += sizeof(float);
  unsigned* cnt  = (unsigned*)(ws + off); off += sizeof(unsigned);

  init_kernel<<<(B + 255) / 256, 256, 0, stream>>>(dap, nmin, smin, flags, sum, cnt, B);
  normalize_kernel<<<(B + 7) / 8, 256, 0, stream>>>(emb, xhi, xlo, B);

  const int nblocks = (B / 128) * SPLIT;      // 8 waves/block share the B panel
  pass1_kernel<<<nblocks, 256, 0, stream>>>(xhi, xlo, labels, dap, nmin, flags, B);
  pass2_kernel<<<nblocks, 256, 0, stream>>>(xhi, xlo, labels, dap, smin, flags, B);

  finalize_kernel<<<(B + 255) / 256, 256, 0, stream>>>(dap, nmin, smin, flags, sum, cnt, B);
  writeout_kernel<<<1, 32, 0, stream>>>(sum, cnt, out);
}
